// SVDHead_47012712022641
// MI455X (gfx1250) — matrix operand order repaired and checked by reading, not compiled, not run on hardware
//
#include <hip/hip_runtime.h>

// ---------------------------------------------------------------- constants
typedef __attribute__((ext_vector_type(16))) __bf16 v16bf;
typedef __attribute__((ext_vector_type(8)))  float  v8f;

constexpr int Bx  = 2, Nn = 2048, Mm = 2048, Cc = 512, Kx = 16, K1x = 8;
constexpr int H1x = 64, H2x = 128;
constexpr int KEY = 1024;                 // N//2 = 1024 (>=900)
constexpr int NT  = Nn / 16;              // 128 n-tiles
constexpr int MT  = Mm / 16;              // 128 m-tiles
constexpr int KC  = Cc / 32;              // 16 k-chunks of 32
constexpr int MGRP = 4;                   // m-tiles per wave in GEMM

// output layout (floats), concatenated in reference return order
constexpr int OFF_R    = 0;
constexpr int OFF_T    = OFF_R   + Bx * 9;            // 18
constexpr int OFF_SKP  = OFF_T   + Bx * 3;            // 24
constexpr int OFF_TKP  = OFF_SKP + Bx * 3 * KEY;      // 6168
constexpr int OFF_SKK  = OFF_TKP + Bx * 3 * KEY;      // 12312
constexpr int OFF_TKK  = OFF_SKK + Bx * 3 * KEY * Kx; // 110616
constexpr int OFF_LOSS = OFF_TKK + Bx * 3 * KEY * Kx; // 208920

static __device__ __forceinline__ unsigned short f2bf(float f) {
  unsigned int u = __float_as_uint(f);
  unsigned int r = u + 0x7FFFu + ((u >> 16) & 1u);   // RNE
  return (unsigned short)(r >> 16);
}

// ---------------------------------------------------------------- pack A
// A fragment (16x32 bf16, M=n rows, K=c): lane l: row = l&15,
// element e -> K = (l<16?0:16) + e
__global__ void svdh_pack_a(const float* __restrict__ emb,
                            unsigned short* __restrict__ ap) {
  int gid  = blockIdx.x * blockDim.x + threadIdx.x;
  int lane = gid & 31;
  int tile = gid >> 5;                       // (b*NT + nt)*KC + kc
  if (tile >= Bx * NT * KC) return;
  int kc = tile % KC; int t2 = tile / KC;
  int nt = t2 % NT;   int b  = t2 / NT;
  int n  = nt * 16 + (lane & 15);
  int kb = kc * 32 + ((lane & 16) ? 16 : 0);
  const float* base = emb + (size_t)b * Cc * Nn + n;
  unsigned short* dst = ap + (size_t)tile * 512 + lane * 16;
#pragma unroll
  for (int e = 0; e < 16; e++) {
    int ko = kb + e;
    dst[e] = f2bf(base[(size_t)ko * Nn]);
  }
}

// ---------------------------------------------------------------- pack B
// B fragment (32x16 bf16, K=c rows, N=m cols): lane l: col = l&15,
// element e -> K = (l<16?0:16) + e
__global__ void svdh_pack_b(const float* __restrict__ emb,
                            unsigned short* __restrict__ bp) {
  int gid  = blockIdx.x * blockDim.x + threadIdx.x;
  int lane = gid & 31;
  int tile = gid >> 5;                       // (b*MT + mt)*KC + kc
  if (tile >= Bx * MT * KC) return;
  int kc = tile % KC; int t2 = tile / KC;
  int mt = t2 % MT;   int b  = t2 / MT;
  int m  = mt * 16 + (lane & 15);
  int kb = kc * 32 + ((lane & 16) ? 16 : 0);
  const float* base = emb + (size_t)b * Cc * Mm + m;
  unsigned short* dst = bp + (size_t)tile * 512 + lane * 16;
#pragma unroll
  for (int e = 0; e < 16; e++)
    dst[e] = f2bf(base[(size_t)(kb + e) * Mm]);
}

// ---------------------------------------------------------------- sq norms
__global__ void svdh_sqnorm(const float* __restrict__ emb, float* __restrict__ out,
                            int npts) {
  int g = blockIdx.x * blockDim.x + threadIdx.x;
  if (g >= Bx * npts) return;
  int b = g / npts, n = g % npts;
  const float* base = emb + (size_t)b * Cc * npts + n;
  float s = 0.f;
  for (int c = 0; c < Cc; c++) { float v = base[(size_t)c * npts]; s += v * v; }
  out[g] = s;
}

// ---------------------------------------------------------------- GEMM (WMMA)
// one wave per 16x64 strip of dist; dist = xx + yy - 2 * (A^T B)
// Software-pipelined: ping-pong register double buffering of A and B fragments
// so the next k-step's 10 global_load_b128 are in flight during this step's
// 4 WMMAs (avoids the s_wait_loadcnt 0 -> wmma serialization).
__global__ __launch_bounds__(32)
void svdh_gemm_dist(const unsigned short* __restrict__ ap,
                    const unsigned short* __restrict__ bp,
                    const float* __restrict__ xx, const float* __restrict__ yy,
                    float* __restrict__ dist) {
  int lane = threadIdx.x;
  int blk  = blockIdx.x;                 // (b, nt, mg)
  int mg = blk % (MT / MGRP); int t = blk / (MT / MGRP);
  int nt = t % NT;            int b = t / NT;

  const unsigned short* abase =
      ap + ((size_t)(b * NT + nt) * KC) * 512 + lane * 16;
  const unsigned short* bbase =
      bp + ((size_t)(b * MT + mg * MGRP) * KC) * 512 + lane * 16;

  v8f acc[MGRP] = {};

  // prologue: buffer A = step 0
  v16bf aA = *(const v16bf*)(abase);
  v16bf bA0 = *(const v16bf*)(bbase + (size_t)0 * KC * 512);
  v16bf bA1 = *(const v16bf*)(bbase + (size_t)1 * KC * 512);
  v16bf bA2 = *(const v16bf*)(bbase + (size_t)2 * KC * 512);
  v16bf bA3 = *(const v16bf*)(bbase + (size_t)3 * KC * 512);
  v16bf aB, bB0, bB1, bB2, bB3;

#pragma unroll 1
  for (int kc = 0; kc < KC; kc += 2) {
    // prefetch step kc+1 into buffer B
    {
      const unsigned short* an = abase + (size_t)(kc + 1) * 512;
      const unsigned short* bn = bbase + (size_t)(kc + 1) * 512;
      aB  = *(const v16bf*)(an);
      bB0 = *(const v16bf*)(bn + (size_t)0 * KC * 512);
      bB1 = *(const v16bf*)(bn + (size_t)1 * KC * 512);
      bB2 = *(const v16bf*)(bn + (size_t)2 * KC * 512);
      bB3 = *(const v16bf*)(bn + (size_t)3 * KC * 512);
    }
    // compute step kc from buffer A
    acc[0] = __builtin_amdgcn_wmma_f32_16x16x32_bf16(false, aA, false, bA0,
                                                     (short)0, acc[0], false, false);
    acc[1] = __builtin_amdgcn_wmma_f32_16x16x32_bf16(false, aA, false, bA1,
                                                     (short)0, acc[1], false, false);
    acc[2] = __builtin_amdgcn_wmma_f32_16x16x32_bf16(false, aA, false, bA2,
                                                     (short)0, acc[2], false, false);
    acc[3] = __builtin_amdgcn_wmma_f32_16x16x32_bf16(false, aA, false, bA3,
                                                     (short)0, acc[3], false, false);
    // prefetch step kc+2 into buffer A
    if (kc + 2 < KC) {
      const unsigned short* an = abase + (size_t)(kc + 2) * 512;
      const unsigned short* bn = bbase + (size_t)(kc + 2) * 512;
      aA  = *(const v16bf*)(an);
      bA0 = *(const v16bf*)(bn + (size_t)0 * KC * 512);
      bA1 = *(const v16bf*)(bn + (size_t)1 * KC * 512);
      bA2 = *(const v16bf*)(bn + (size_t)2 * KC * 512);
      bA3 = *(const v16bf*)(bn + (size_t)3 * KC * 512);
      __builtin_prefetch(abase + (size_t)(kc + 3) * 512, 0, 3);
    }
    // compute step kc+1 from buffer B
    acc[0] = __builtin_amdgcn_wmma_f32_16x16x32_bf16(false, aB, false, bB0,
                                                     (short)0, acc[0], false, false);
    acc[1] = __builtin_amdgcn_wmma_f32_16x16x32_bf16(false, aB, false, bB1,
                                                     (short)0, acc[1], false, false);
    acc[2] = __builtin_amdgcn_wmma_f32_16x16x32_bf16(false, aB, false, bB2,
                                                     (short)0, acc[2], false, false);
    acc[3] = __builtin_amdgcn_wmma_f32_16x16x32_bf16(false, aB, false, bB3,
                                                     (short)0, acc[3], false, false);
  }

  int rbase = (lane >> 4) * 8;    // C/D layout: VGPR v -> M = v (+8 for hi lanes)
  int col   = lane & 15;
#pragma unroll
  for (int g2 = 0; g2 < MGRP; g2++) {
    int m  = (mg * MGRP + g2) * 16 + col;
    float yv = yy[b * Mm + m];
#pragma unroll
    for (int v = 0; v < 8; v++) {
      int n2 = nt * 16 + rbase + v;
      dist[((size_t)b * Nn + n2) * Mm + m] =
          xx[b * Nn + n2] + yv - 2.0f * acc[g2][v];
    }
  }
}

// ---------------------------------------------------------------- softmax(-x) per row
__global__ void svdh_softmax_neg(const float* __restrict__ in,
                                 float* __restrict__ out) {
  __shared__ float red[256];
  int row = blockIdx.x, tid = threadIdx.x;
  const float* r = in + (size_t)row * Mm;
  float* o = out + (size_t)row * Mm;
  float mn = 3.4e38f;
  for (int m = tid; m < Mm; m += 256) mn = fminf(mn, r[m]);
  red[tid] = mn; __syncthreads();
  for (int off = 128; off; off >>= 1) {
    if (tid < off) red[tid] = fminf(red[tid], red[tid + off]);
    __syncthreads();
  }
  mn = red[0]; __syncthreads();
  float s = 0.f;
  for (int m = tid; m < Mm; m += 256) s += __expf(mn - r[m]);
  red[tid] = s; __syncthreads();
  for (int off = 128; off; off >>= 1) {
    if (tid < off) red[tid] += red[tid + off];
    __syncthreads();
  }
  float inv = 1.0f / red[0];
  for (int m = tid; m < Mm; m += 256) o[m] = __expf(mn - r[m]) * inv;
}

// ---------------------------------------------------------------- T[b,n,m] = sum_{i=1..7} scores[b, idx1[b,n,i], m]
__global__ void svdh_tsum(const float* __restrict__ scores,
                          const int* __restrict__ idx1,
                          float* __restrict__ T) {
  __shared__ int g[K1x];
  int row = blockIdx.x, tid = threadIdx.x;
  int b = row / Nn, n = row % Nn;
  if (tid < K1x) g[tid] = idx1[((size_t)b * Nn + n) * K1x + tid];
  __syncthreads();
  const float* sb = scores + (size_t)b * Nn * Mm;
  float* To = T + (size_t)row * Mm;
  for (int m = tid; m < Mm; m += 256) {
    float s = 0.f;
#pragma unroll
    for (int i = 1; i < K1x; i++) s += sb[(size_t)g[i] * Mm + m];
    To[m] = s;
  }
}

// ---------------------------------------------------------------- refined = exp(1 - S/7) * dist
__global__ void svdh_refined(const float* __restrict__ dist,
                             const float* __restrict__ T,
                             const int* __restrict__ idx2,
                             float* __restrict__ refined) {
  int row = blockIdx.x, tid = threadIdx.x;
  int b = row / Nn;
  const float* Tr = T + (size_t)row * Mm;
  const float* dr = dist + (size_t)row * Mm;
  float* rr = refined + (size_t)row * Mm;
  const int* i2 = idx2 + (size_t)b * Mm * K1x;
  for (int m = tid; m < Mm; m += 256) {
    float s = 0.f;
#pragma unroll
    for (int j = 1; j < K1x; j++) s += Tr[i2[(size_t)m * K1x + j]];
    float sk = 1.0f - s * (1.0f / (K1x - 1));       // NN_MARGIN - S/(k1-1)
    rr[m] = __expf(sk) * dr[m];
  }
}

// ---------------------------------------------------------------- rmm = softmax(-refined) in place + argmax idx
__global__ void svdh_rmm(float* __restrict__ buf, int* __restrict__ itc) {
  __shared__ float rv[256];
  __shared__ int   ri[256];
  int row = blockIdx.x, tid = threadIdx.x;
  float* r = buf + (size_t)row * Mm;
  float bv = 3.4e38f; int bi = 0;
  for (int m = tid; m < Mm; m += 256) {
    float v = r[m];
    if (v < bv || (v == bv && m < bi)) { bv = v; bi = m; }
  }
  rv[tid] = bv; ri[tid] = bi; __syncthreads();
  for (int off = 128; off; off >>= 1) {
    if (tid < off) {
      float v = rv[tid + off]; int i = ri[tid + off];
      if (v < rv[tid] || (v == rv[tid] && i < ri[tid])) { rv[tid] = v; ri[tid] = i; }
    }
    __syncthreads();
  }
  float mn = rv[0];
  if (tid == 0) itc[row] = ri[0];
  __syncthreads();
  float s = 0.f;
  for (int m = tid; m < Mm; m += 256) s += __expf(mn - r[m]);
  rv[tid] = s; __syncthreads();
  for (int off = 128; off; off >>= 1) {
    if (tid < off) rv[tid] += rv[tid + off];
    __syncthreads();
  }
  float inv = 1.0f / rv[0];
  for (int m = tid; m < Mm; m += 256) r[m] = __expf(mn - r[m]) * inv;
}

// ---------------------------------------------------------------- src_corr[b,c,n] = sum_m tgt[b,c,m]*rmm[b,n,m]
__global__ void svdh_corr(const float* __restrict__ tgt,
                          const float* __restrict__ rmm,
                          float* __restrict__ corr) {
  __shared__ float red[256];
  int row = blockIdx.x, tid = threadIdx.x;
  int b = row / Nn, n = row % Nn;
  const float* rr = rmm + (size_t)row * Mm;
  const float* tb = tgt + (size_t)b * 3 * Mm;
  float a0 = 0.f, a1 = 0.f, a2 = 0.f;
  for (int m = tid; m < Mm; m += 256) {
    float w = rr[m];
    a0 += w * tb[m]; a1 += w * tb[Mm + m]; a2 += w * tb[2 * Mm + m];
  }
  float acc[3] = {a0, a1, a2};
  for (int c = 0; c < 3; c++) {
    red[tid] = acc[c]; __syncthreads();
    for (int off = 128; off; off >>= 1) {
      if (tid < off) red[tid] += red[tid + off];
      __syncthreads();
    }
    if (tid == 0) corr[(size_t)(b * 3 + c) * Nn + n] = red[0];
    __syncthreads();
  }
}

// ---------------------------------------------------------------- discriminator MLP -> per-point score
__global__ __launch_bounds__(128)
void svdh_disc(const float* __restrict__ src, const float* __restrict__ src_knn,
               const float* __restrict__ corr, const int* __restrict__ src_idx,
               const float* __restrict__ W1, const float* __restrict__ b1,
               const float* __restrict__ W2, const float* __restrict__ b2,
               const float* __restrict__ W3, const float* __restrict__ b3,
               float* __restrict__ sval, float* __restrict__ knnd) {
  __shared__ float sW1[H1x * 6], sB1[H1x], sW2[H2x * H1x], sB2[H2x], sW3[H2x];
  __shared__ float sB3;
  __shared__ float sred[128];
  int tid = threadIdx.x;
  for (int i = tid; i < H1x * 6; i += 128) sW1[i] = W1[i];
  for (int i = tid; i < H1x; i += 128) sB1[i] = b1[i];
  for (int i = tid; i < H2x * H1x; i += 128) sW2[i] = W2[i];
  for (int i = tid; i < H2x; i += 128) sB2[i] = b2[i];
  for (int i = tid; i < H2x; i += 128) sW3[i] = W3[i];
  if (tid == 0) sB3 = b3[0];
  __syncthreads();

  int chunk = blockIdx.x % (Nn / 8);
  int b     = blockIdx.x / (Nn / 8);
  int nl = tid >> 4, kk = tid & 15;
  int n  = chunk * 8 + nl;

  int m = src_idx[((size_t)b * Nn + n) * Kx + kk];
  const float* cb = corr + (size_t)b * 3 * Nn;
  const float* sb = src  + (size_t)b * 3 * Nn;
  float f[6];
#pragma unroll
  for (int c = 0; c < 3; c++) {
    float kd = cb[c * Nn + n] - cb[c * Nn + m];
    f[c] = kd;
    knnd[(((size_t)b * Nn + n) * Kx + kk) * 3 + c] = kd;
    f[3 + c] = sb[c * Nn + n] - src_knn[(((size_t)b * Nn + n) * Kx + kk) * 3 + c];
  }
  float h1v[H1x];
#pragma unroll 4
  for (int d = 0; d < H1x; d++) {
    float a = sB1[d];
#pragma unroll
    for (int c = 0; c < 6; c++) a += sW1[d * 6 + c] * f[c];
    h1v[d] = fmaxf(a, 0.f);
  }
  float s = sB3;
  for (int e = 0; e < H2x; e++) {
    float a = sB2[e];
#pragma unroll 8
    for (int d = 0; d < H1x; d++) a += sW2[e * H1x + d] * h1v[d];
    s += sW3[e] * fmaxf(a, 0.f);
  }
  sred[tid] = s; __syncthreads();
  for (int off = 8; off; off >>= 1) {
    if (kk < off) sred[tid] = fmaxf(sred[tid], sred[tid + off]);
    __syncthreads();
  }
  if (kk == 0) sval[(size_t)b * Nn + n] = sred[tid];
}

// ---------------------------------------------------------------- softmax over n (weight)
__global__ void svdh_wsoftmax(const float* __restrict__ sval,
                              float* __restrict__ w) {
  __shared__ float red[256];
  int b = blockIdx.x, tid = threadIdx.x;
  const float* r = sval + (size_t)b * Nn;
  float* o = w + (size_t)b * Nn;
  float mx = -3.4e38f;
  for (int n = tid; n < Nn; n += 256) mx = fmaxf(mx, r[n]);
  red[tid] = mx; __syncthreads();
  for (int off = 128; off; off >>= 1) {
    if (tid < off) red[tid] = fmaxf(red[tid], red[tid + off]);
    __syncthreads();
  }
  mx = red[0]; __syncthreads();
  float s = 0.f;
  for (int n = tid; n < Nn; n += 256) s += __expf(r[n] - mx);
  red[tid] = s; __syncthreads();
  for (int off = 128; off; off >>= 1) {
    if (tid < off) red[tid] += red[tid + off];
    __syncthreads();
  }
  float inv = 1.0f / red[0];
  for (int n = tid; n < Nn; n += 256) o[n] = __expf(r[n] - mx) * inv;
}

// ---------------------------------------------------------------- weighted centroids
__global__ void svdh_rigid1(const float* __restrict__ src,
                            const float* __restrict__ corr,
                            const float* __restrict__ w,
                            float* __restrict__ csct) {
  __shared__ float red[256];
  int b = blockIdx.x, tid = threadIdx.x;
  float a[6] = {0, 0, 0, 0, 0, 0};
  for (int n = tid; n < Nn; n += 256) {
    float wv = w[(size_t)b * Nn + n];
#pragma unroll
    for (int c = 0; c < 3; c++) {
      a[c]     += src[(size_t)(b * 3 + c) * Nn + n] * wv;
      a[3 + c] += corr[(size_t)(b * 3 + c) * Nn + n] * wv;
    }
  }
  for (int i = 0; i < 6; i++) {
    red[tid] = a[i]; __syncthreads();
    for (int off = 128; off; off >>= 1) {
      if (tid < off) red[tid] += red[tid + off];
      __syncthreads();
    }
    if (tid == 0) csct[b * 6 + i] = red[0];
    __syncthreads();
  }
}

// ---------------------------------------------------------------- weighted covariance H
__global__ void svdh_rigid2(const float* __restrict__ src,
                            const float* __restrict__ corr,
                            const float* __restrict__ w,
                            const float* __restrict__ csct,
                            float* __restrict__ Hm) {
  __shared__ float red[256];
  int b = blockIdx.x, tid = threadIdx.x;
  float cs[3], ct[3];
#pragma unroll
  for (int c = 0; c < 3; c++) { cs[c] = csct[b * 6 + c]; ct[c] = csct[b * 6 + 3 + c]; }
  float h[9] = {0, 0, 0, 0, 0, 0, 0, 0, 0};
  for (int n = tid; n < Nn; n += 256) {
    float wv = w[(size_t)b * Nn + n];
    float sx[3], cy[3];
#pragma unroll
    for (int c = 0; c < 3; c++) {
      sx[c] = (src[(size_t)(b * 3 + c) * Nn + n] - cs[c]) * wv;
      cy[c] = corr[(size_t)(b * 3 + c) * Nn + n] - ct[c];
    }
#pragma unroll
    for (int i = 0; i < 3; i++)
#pragma unroll
      for (int j = 0; j < 3; j++) h[i * 3 + j] += sx[i] * cy[j];
  }
  for (int i = 0; i < 9; i++) {
    red[tid] = h[i]; __syncthreads();
    for (int off = 128; off; off >>= 1) {
      if (tid < off) red[tid] += red[tid + off];
      __syncthreads();
    }
    if (tid == 0) Hm[b * 9 + i] = red[0];
    __syncthreads();
  }
}

// ---------------------------------------------------------------- 3x3 SVD (one lane per batch)
static __device__ __forceinline__ float det3(const float* M) {
  return M[0] * (M[4] * M[8] - M[5] * M[7]) - M[1] * (M[3] * M[8] - M[5] * M[6]) +
         M[2] * (M[3] * M[7] - M[4] * M[6]);
}

__global__ void svdh_svd(const float* __restrict__ Hm,
                         const float* __restrict__ csct,
                         float* __restrict__ out, float* __restrict__ Rws) {
  int b = threadIdx.x;
  if (b >= Bx) return;
  float H[9];
#pragma unroll
  for (int i = 0; i < 9; i++) H[i] = Hm[b * 9 + i];
  // A = H^T H
  float A[3][3], V[3][3];
#pragma unroll
  for (int i = 0; i < 3; i++)
#pragma unroll
    for (int j = 0; j < 3; j++) {
      float s = 0.f;
      for (int k = 0; k < 3; k++) s += H[k * 3 + i] * H[k * 3 + j];
      A[i][j] = s; V[i][j] = (i == j) ? 1.f : 0.f;
    }
  const int PP[3] = {0, 0, 1}, QQ[3] = {1, 2, 2};
  for (int it = 0; it < 30; it++) {
    int p = PP[it % 3], q = QQ[it % 3];
    float apq = A[p][q];
    if (fabsf(apq) < 1e-30f) continue;
    float tau = (A[q][q] - A[p][p]) / (2.f * apq);
    float tt  = ((tau >= 0.f) ? 1.f : -1.f) / (fabsf(tau) + sqrtf(1.f + tau * tau));
    float c = 1.f / sqrtf(1.f + tt * tt), s = tt * c;
    for (int k = 0; k < 3; k++) {
      float akp = A[k][p], akq = A[k][q];
      A[k][p] = c * akp - s * akq; A[k][q] = s * akp + c * akq;
    }
    for (int k = 0; k < 3; k++) {
      float apk = A[p][k], aqk = A[q][k];
      A[p][k] = c * apk - s * aqk; A[q][k] = s * apk + c * aqk;
    }
    for (int k = 0; k < 3; k++) {
      float vkp = V[k][p], vkq = V[k][q];
      V[k][p] = c * vkp - s * vkq; V[k][q] = s * vkp + c * vkq;
    }
  }
  float lam[3] = {A[0][0], A[1][1], A[2][2]};
  // sort descending, permute V columns
  for (int a = 0; a < 2; a++)
    for (int bb = a + 1; bb < 3; bb++)
      if (lam[a] < lam[bb]) {
        float t = lam[a]; lam[a] = lam[bb]; lam[bb] = t;
        for (int k = 0; k < 3; k++) { float v = V[k][a]; V[k][a] = V[k][bb]; V[k][bb] = v; }
      }
  float U[3][3];
  for (int i = 0; i < 3; i++) {
    float u[3];
    for (int k = 0; k < 3; k++) {
      float s = 0.f;
      for (int j = 0; j < 3; j++) s += H[k * 3 + j] * V[j][i];
      u[k] = s;
    }
    float sig = sqrtf(fmaxf(lam[i], 0.f));
    if (sig > 1e-8f) {
      for (int k = 0; k < 3; k++) U[k][i] = u[k] / sig;
    } else if (i == 2) {
      U[0][2] = U[1][0] * U[2][1] - U[2][0] * U[1][1];
      U[1][2] = U[2][0] * U[0][1] - U[0][0] * U[2][1];
      U[2][2] = U[0][0] * U[1][1] - U[1][0] * U[0][1];
    } else {
      float nrm = sqrtf(u[0] * u[0] + u[1] * u[1] + u[2] * u[2]) + 1e-20f;
      for (int k = 0; k < 3; k++) U[k][i] = u[k] / nrm;
    }
  }
  float Vf[9], Uf[9];
  for (int i = 0; i < 3; i++)
    for (int j = 0; j < 3; j++) { Vf[i * 3 + j] = V[i][j]; Uf[i * 3 + j] = U[i][j]; }
  float d = det3(Vf) * det3(Uf);
  float R[9];
  for (int i = 0; i < 3; i++)
    for (int k = 0; k < 3; k++)
      R[i * 3 + k] = V[i][0] * U[k][0] + V[i][1] * U[k][1] + d * V[i][2] * U[k][2];
  float cs[3], ct[3], t[3];
  for (int c = 0; c < 3; c++) { cs[c] = csct[b * 6 + c]; ct[c] = csct[b * 6 + 3 + c]; }
  for (int i = 0; i < 3; i++)
    t[i] = ct[i] - (R[i * 3 + 0] * cs[0] + R[i * 3 + 1] * cs[1] + R[i * 3 + 2] * cs[2]);
  for (int i = 0; i < 9; i++) { out[OFF_R + b * 9 + i] = R[i]; Rws[b * 9 + i] = R[i]; }
  for (int i = 0; i < 3; i++) out[OFF_T + b * 3 + i] = t[i];
}

// ---------------------------------------------------------------- top-1024 per batch (bitonic sort of 2048)
__global__ __launch_bounds__(1024)
void svdh_topk(const float* __restrict__ w, int* __restrict__ topi) {
  __shared__ float wv[2048];
  __shared__ int   wi[2048];
  int b = blockIdx.x, tid = threadIdx.x;
  wv[tid] = w[(size_t)b * Nn + tid];           wi[tid] = tid;
  wv[tid + 1024] = w[(size_t)b * Nn + tid + 1024]; wi[tid + 1024] = tid + 1024;
  __syncthreads();
  for (int k = 2; k <= 2048; k <<= 1) {
    for (int j = k >> 1; j > 0; j >>= 1) {
#pragma unroll
      for (int rep = 0; rep < 2; rep++) {
        int i = tid + rep * 1024;
        int ixj = i ^ j;
        if (ixj > i) {
          // "first in descending order" comparator; ties -> smaller index first
          bool f = (wv[i] > wv[ixj]) || (wv[i] == wv[ixj] && wi[i] < wi[ixj]);
          bool doswap = ((i & k) == 0) ? (!f) : f;
          if (doswap) {
            float tv = wv[i]; wv[i] = wv[ixj]; wv[ixj] = tv;
            int ti = wi[i]; wi[i] = wi[ixj]; wi[ixj] = ti;
          }
        }
      }
      __syncthreads();
    }
  }
  topi[(size_t)b * KEY + tid] = wi[tid];
}

// ---------------------------------------------------------------- keypoints + per-row loss
__global__ void svdh_keyp(const float* __restrict__ src,
                          const float* __restrict__ corr,
                          const int* __restrict__ topi,
                          const float* __restrict__ rmm,
                          const int* __restrict__ itc,
                          float* __restrict__ out,
                          float* __restrict__ lossbuf) {
  int g = blockIdx.x * blockDim.x + threadIdx.x;
  if (g >= Bx * KEY) return;
  int b = g / KEY, j = g % KEY;
  int n = topi[(size_t)b * KEY + j];
#pragma unroll
  for (int c = 0; c < 3; c++) {
    out[OFF_SKP + (size_t)(b * 3 + c) * KEY + j] = src[(size_t)(b * 3 + c) * Nn + n];
    out[OFF_TKP + (size_t)(b * 3 + c) * KEY + j] = corr[(size_t)(b * 3 + c) * Nn + n];
  }
  int m = itc[(size_t)b * Nn + n];
  float p = rmm[((size_t)b * Nn + n) * Mm + m];
  lossbuf[g] = -logf(p + 1e-15f);
}

__global__ void svdh_loss_reduce(const float* __restrict__ lossbuf,
                                 float* __restrict__ out) {
  __shared__ float red[256];
  int tid = threadIdx.x;
  float s = 0.f;
  for (int g = tid; g < Bx * KEY; g += 256) s += lossbuf[g];
  red[tid] = s; __syncthreads();
  for (int off = 128; off; off >>= 1) {
    if (tid < off) red[tid] += red[tid + off];
    __syncthreads();
  }
  if (tid == 0) out[OFF_LOSS] = red[0] * (1.0f / (Bx * KEY));
}

// ---------------------------------------------------------------- knn outputs
__global__ void svdh_knnout(const float* __restrict__ src,
                            const int* __restrict__ src_idx,
                            const int* __restrict__ topi,
                            const float* __restrict__ knnd,
                            const float* __restrict__ Rws,
                            float* __restrict__ out) {
  int g = blockIdx.x * blockDim.x + threadIdx.x;
  if (g >= Bx * KEY * Kx) return;
  int kk = g % Kx; int t = g / Kx;
  int j = t % KEY; int b = t / KEY;
  int n = topi[(size_t)b * KEY + j];
#pragma unroll
  for (int c = 0; c < 3; c++)
    out[OFF_TKK + ((size_t)(b * 3 + c) * KEY + j) * Kx + kk] =
        knnd[(((size_t)b * Nn + n) * Kx + kk) * 3 + c];
  int m = src_idx[((size_t)b * Nn + n) * Kx + kk];
  float d0 = src[(size_t)(b * 3 + 0) * Nn + n] - src[(size_t)(b * 3 + 0) * Nn + m];
  float d1 = src[(size_t)(b * 3 + 1) * Nn + n] - src[(size_t)(b * 3 + 1) * Nn + m];
  float d2 = src[(size_t)(b * 3 + 2) * Nn + n] - src[(size_t)(b * 3 + 2) * Nn + m];
  const float* R = Rws + (size_t)b * 9;
#pragma unroll
  for (int c = 0; c < 3; c++)
    out[OFF_SKK + ((size_t)(b * 3 + c) * KEY + j) * Kx + kk] =
        R[c * 3 + 0] * d0 + R[c * 3 + 1] * d1 + R[c * 3 + 2] * d2;
}

// ================================================================ host launch
extern "C" void kernel_launch(void* const* d_in, const int* in_sizes, int n_in,
                              void* d_out, int out_size, void* d_ws, size_t ws_size,
                              hipStream_t stream) {
  const float* src     = (const float*)d_in[0];
  const float* tgt     = (const float*)d_in[1];
  const float* semb    = (const float*)d_in[2];
  const float* temb    = (const float*)d_in[3];
  const float* src_knn = (const float*)d_in[4];
  const float* W1 = (const float*)d_in[6];
  const float* b1 = (const float*)d_in[7];
  const float* W2 = (const float*)d_in[8];
  const float* b2 = (const float*)d_in[9];
  const float* W3 = (const float*)d_in[10];
  const float* b3 = (const float*)d_in[11];
  const int* src_idx  = (const int*)d_in[12];
  const int* src_idx1 = (const int*)d_in[13];
  const int* idx2     = (const int*)d_in[14];
  float* out = (float*)d_out;

  char* ws = (char*)d_ws;
  size_t o = 0;
  unsigned short* apack = (unsigned short*)(ws + o); o += (size_t)Bx * NT * KC * 512 * 2;
  unsigned short* bpack = (unsigned short*)(ws + o); o += (size_t)Bx * MT * KC * 512 * 2;
  float* xx   = (float*)(ws + o); o += (size_t)Bx * Nn * 4;
  float* yy   = (float*)(ws + o); o += (size_t)Bx * Mm * 4;
  float* dist = (float*)(ws + o); o += (size_t)Bx * Nn * Mm * 4;
  float* scor = (float*)(ws + o); o += (size_t)Bx * Nn * Mm * 4;  // scores -> refined -> rmm
  float* Tbuf = (float*)(ws + o); o += (size_t)Bx * Nn * Mm * 4;
  float* corr = (float*)(ws + o); o += (size_t)Bx * 3 * Nn * 4;
  float* knnd = (float*)(ws + o); o += (size_t)Bx * Nn * Kx * 3 * 4;
  float* sval = (float*)(ws + o); o += (size_t)Bx * Nn * 4;
  float* wgt  = (float*)(ws + o); o += (size_t)Bx * Nn * 4;
  int*   itc  = (int*)(ws + o);   o += (size_t)Bx * Nn * 4;
  int*   topi = (int*)(ws + o);   o += (size_t)Bx * KEY * 4;
  float* csct = (float*)(ws + o); o += 256;
  float* Hm   = (float*)(ws + o); o += 256;
  float* Rws  = (float*)(ws + o); o += 256;
  float* lossbuf = (float*)(ws + o); o += (size_t)Bx * KEY * 4;

  // 1) pack embeddings to bf16 WMMA fragment layout + squared norms
  {
    int tilesA = Bx * NT * KC, tilesB = Bx * MT * KC;
    svdh_pack_a<<<dim3((tilesA * 32 + 255) / 256), dim3(256), 0, stream>>>(semb, apack);
    svdh_pack_b<<<dim3((tilesB * 32 + 255) / 256), dim3(256), 0, stream>>>(temb, bpack);
    svdh_sqnorm<<<dim3((Bx * Nn + 255) / 256), dim3(256), 0, stream>>>(semb, xx, Nn);
    svdh_sqnorm<<<dim3((Bx * Mm + 255) / 256), dim3(256), 0, stream>>>(temb, yy, Mm);
  }
  // 2) WMMA GEMM -> dist (software pipelined)
  svdh_gemm_dist<<<dim3(Bx * NT * (MT / MGRP)), dim3(32), 0, stream>>>(
      apack, bpack, xx, yy, dist);
  // 3) scores = softmax(-dist)
  svdh_softmax_neg<<<dim3(Bx * Nn), dim3(256), 0, stream>>>(dist, scor);
  // 4) T
  svdh_tsum<<<dim3(Bx * Nn), dim3(256), 0, stream>>>(scor, src_idx1, Tbuf);
  // 5) refined (overwrites scores buffer)
  svdh_refined<<<dim3(Bx * Nn), dim3(256), 0, stream>>>(dist, Tbuf, idx2, scor);
  // 6) rmm = softmax(-refined) in place, + argmax index
  svdh_rmm<<<dim3(Bx * Nn), dim3(256), 0, stream>>>(scor, itc);
  // 7) src_corr
  svdh_corr<<<dim3(Bx * Nn), dim3(256), 0, stream>>>(tgt, scor, corr);
  // 8) discriminator scores + knn_distance
  svdh_disc<<<dim3(Bx * (Nn / 8)), dim3(128), 0, stream>>>(
      src, src_knn, corr, src_idx, W1, b1, W2, b2, W3, b3, sval, knnd);
  // 9) weight = softmax over n
  svdh_wsoftmax<<<dim3(Bx), dim3(256), 0, stream>>>(sval, wgt);
  // 10) rigid transform
  svdh_rigid1<<<dim3(Bx), dim3(256), 0, stream>>>(src, corr, wgt, csct);
  svdh_rigid2<<<dim3(Bx), dim3(256), 0, stream>>>(src, corr, wgt, csct, Hm);
  svdh_svd<<<dim3(1), dim3(32), 0, stream>>>(Hm, csct, out, Rws);
  // 11) top-1024 keypoints
  svdh_topk<<<dim3(Bx), dim3(1024), 0, stream>>>(wgt, topi);
  // 12) keypoint gathers + loss
  svdh_keyp<<<dim3((Bx * KEY + 255) / 256), dim3(256), 0, stream>>>(
      src, corr, topi, scor, itc, out, lossbuf);
  svdh_loss_reduce<<<dim3(1), dim3(256), 0, stream>>>(lossbuf, out);
  // 13) knn outputs
  svdh_knnout<<<dim3((Bx * KEY * Kx + 255) / 256), dim3(256), 0, stream>>>(
      src, src_idx, topi, knnd, Rws, out);
}
